// TransformerCN_60009283060271
// MI455X (gfx1250) — matrix operand order repaired and checked by reading, not compiled, not run on hardware
//
#include <hip/hip_runtime.h>
#include <hip/hip_bf16.h>
#include <math.h>

// Problem constants (match reference)
#define NN   20000     // nodes
#define NE   320000    // edges
#define NG   256       // graphs
#define DIN  64        // input feature dim
#define HC   256       // heads*hid
#define NH   4         // heads
#define NT   12        // task heads

typedef __attribute__((ext_vector_type(16))) __bf16 v16bf;
typedef __attribute__((ext_vector_type(8)))  float  v8f;

// ---------------------------------------------------------------- utilities

__global__ void fill_f32(float* p, float v, int n) {
    int i = blockIdx.x * blockDim.x + threadIdx.x;
    if (i < n) p[i] = v;
}

__global__ void cvt_f32_bf16(const float* __restrict__ in, __bf16* __restrict__ out, int n) {
    int i = blockIdx.x * blockDim.x + threadIdx.x;
    if (i < n) out[i] = (__bf16)in[i];
}

__global__ void relu_cvt_f32_bf16(const float* __restrict__ in, __bf16* __restrict__ out, int n) {
    int i = blockIdx.x * blockDim.x + threadIdx.x;
    if (i < n) {
        float v = in[i];
        out[i] = (__bf16)fmaxf(v, 0.0f);
    }
}

// Swizzle a row-major f32 weight W[Kdim x HC] into a contiguous-k bf16 B-fragment
// layout: for col-tile nt, k-tile kt, lane L (n=L%16, kh=L/16), vgpr g holds
// the packed pair { W[kt*32+kh*16+2g, nt*16+n], W[kt*32+kh*16+2g+1, ...] }.
// Stored contiguously per lane so the GEMM loads 2x b128 per fragment.
__global__ void swizzle_w(const float* __restrict__ W, __bf16* __restrict__ Wsw, int Kdim) {
    int total = Kdim * HC;
    int i = blockIdx.x * blockDim.x + threadIdx.x;
    if (i >= total) return;
    int KT  = Kdim >> 5;
    int p   = i & 1;
    int g   = (i >> 1) & 7;
    int L   = (i >> 4) & 31;
    int blk = i >> 9;            // nt*KT + kt
    int kt  = blk % KT;
    int nt  = blk / KT;
    int n   = L & 15;
    int kh  = L >> 4;
    int krow = kt * 32 + kh * 16 + 2 * g + p;
    int col  = nt * 16 + n;
    Wsw[i] = (__bf16)W[krow * HC + col];
}

// ---------------------------------------------------------------- WMMA GEMM
// Y[NN x HC] = X[NN x Kdim](bf16) @ W(swizzled bf16) + bias, f32 accumulate.
// Block = 128 threads (4 waves); block handles one 16-row tile; each wave
// produces 4 of the 16 col tiles.  Uses v_wmma_f32_16x16x32_bf16.
__global__ void __launch_bounds__(128)
gemm_wmma_bf16(const __bf16* __restrict__ X, int ldx, int Kdim,
               const __bf16* __restrict__ Wsw, const float* __restrict__ bias,
               float* __restrict__ Y) {
    const int lane = threadIdx.x & 31;
    const int wave = threadIdx.x >> 5;         // 0..3
    const int row0 = blockIdx.x * 16;
    const int m    = lane & 15;
    const int kh   = lane >> 4;
    const int KT   = Kdim >> 5;

    const __bf16* Arow = X + (size_t)(row0 + m) * ldx;

    v8f acc0 = {0.f,0.f,0.f,0.f,0.f,0.f,0.f,0.f};
    v8f acc1 = acc0, acc2 = acc0, acc3 = acc0;

    for (int kt = 0; kt < KT; ++kt) {
        // A fragment: contiguous-K layout (k = kh*16 + s), matching swizzled B; packed consecutive-K bf16 pairs
        union { v16bf v; unsigned u[8]; } af;
        const int kbase = kt * 32 + kh * 16;
        #pragma unroll
        for (int g = 0; g < 8; ++g) {
            int k0 = 2 * g;
            af.u[g] = *(const unsigned*)(Arow + kbase + k0);
        }
        const unsigned* wswu = (const unsigned*)Wsw;
        #pragma unroll
        for (int j = 0; j < 4; ++j) {
            int nt = wave * 4 + j;
            const unsigned* bp = wswu + ((size_t)(nt * KT + kt) * 32 + lane) * 8;
            union { v16bf v; unsigned u[8]; } bf;
            #pragma unroll
            for (int g = 0; g < 8; ++g) bf.u[g] = bp[g];
            v8f c = (j == 0) ? acc0 : (j == 1) ? acc1 : (j == 2) ? acc2 : acc3;
            c = __builtin_amdgcn_wmma_f32_16x16x32_bf16(false, af.v, false, bf.v,
                                                        (short)0, c, false, false);
            if (j == 0) acc0 = c; else if (j == 1) acc1 = c;
            else if (j == 2) acc2 = c; else acc3 = c;
        }
    }

    // C layout: vgpr r, lanes 0-15 -> M=r, lanes 16-31 -> M=8+r; N = lane%16
    #pragma unroll
    for (int j = 0; j < 4; ++j) {
        int col0 = (wave * 4 + j) * 16;
        float b = bias[col0 + m];
        v8f c = (j == 0) ? acc0 : (j == 1) ? acc1 : (j == 2) ? acc2 : acc3;
        #pragma unroll
        for (int r = 0; r < 8; ++r) {
            int row = row0 + r + kh * 8;
            Y[(size_t)row * HC + col0 + m] = c[r] + b;
        }
    }
}

// ---------------------------------------------------------------- edge passes

__device__ __forceinline__ void atomicMaxFloat(float* addr, float val) {
    // monotone int/uint encoding trick; storage initialized to -inf
    if (val >= 0.0f) atomicMax((int*)addr, __float_as_int(val));
    else             atomicMin((unsigned int*)addr, __float_as_uint(val));
}

// Pass 1: alpha[e,h] = q[dst]·(k[src]+e_proj)/8 ; atomic segment-max into amax
__global__ void __launch_bounds__(256)
edge_alpha(const float* __restrict__ Q, const float* __restrict__ Kf,
           const float* __restrict__ eattr, const float* __restrict__ We,
           const int* __restrict__ src, const int* __restrict__ dst,
           float* __restrict__ alpha, float* __restrict__ amax) {
    __shared__ float sWe[4 * HC];
    for (int i = threadIdx.x; i < 4 * HC; i += blockDim.x) sWe[i] = We[i];
    __syncthreads();

    int wid = blockIdx.x * (blockDim.x >> 5) + (threadIdx.x >> 5);
    if (wid >= NE) return;
    int lane = threadIdx.x & 31;
    int s = src[wid], d = dst[wid];
    float e0 = eattr[wid*4+0], e1 = eattr[wid*4+1], e2 = eattr[wid*4+2], e3 = eattr[wid*4+3];
    int c0 = lane * 8;
    float part = 0.f;
    #pragma unroll
    for (int i = 0; i < 8; ++i) {
        int c = c0 + i;
        float ev = e0*sWe[c] + e1*sWe[HC+c] + e2*sWe[2*HC+c] + e3*sWe[3*HC+c];
        part += Q[(size_t)d * HC + c] * (Kf[(size_t)s * HC + c] + ev);
    }
    part += __shfl_xor(part, 1);
    part += __shfl_xor(part, 2);
    part += __shfl_xor(part, 4);
    if ((lane & 7) == 0) {
        int h = lane >> 3;
        float a = part * 0.125f;   // 1/sqrt(64)
        alpha[wid * NH + h] = a;
        atomicMaxFloat(&amax[(size_t)d * NH + h], a);
    }
}

// Pass 2: ex = exp(alpha - amax[dst]); denom += ex
__global__ void edge_exp(const int* __restrict__ dst, const float* __restrict__ amax,
                         float* __restrict__ alpha, float* __restrict__ denom, int n) {
    int i = blockIdx.x * blockDim.x + threadIdx.x;
    if (i >= n) return;
    int e = i >> 2, h = i & 3;
    int d = dst[e];
    float ex = __expf(alpha[i] - amax[(size_t)d * NH + h]);
    alpha[i] = ex;
    atomicAdd(&denom[(size_t)d * NH + h], ex);
}

// Pass 3: out[dst] += (v[src] + e_proj) * (ex / (denom + 1e-16))
__global__ void __launch_bounds__(256)
edge_scatter(const float* __restrict__ V, const float* __restrict__ eattr,
             const float* __restrict__ We, const int* __restrict__ src,
             const int* __restrict__ dst, const float* __restrict__ alpha,
             const float* __restrict__ denom, float* __restrict__ out) {
    __shared__ float sWe[4 * HC];
    for (int i = threadIdx.x; i < 4 * HC; i += blockDim.x) sWe[i] = We[i];
    __syncthreads();

    int wid = blockIdx.x * (blockDim.x >> 5) + (threadIdx.x >> 5);
    if (wid >= NE) return;
    int lane = threadIdx.x & 31;
    int s = src[wid], d = dst[wid];
    int h = lane >> 3;
    float a = alpha[wid * NH + h] / (denom[(size_t)d * NH + h] + 1e-16f);
    float e0 = eattr[wid*4+0], e1 = eattr[wid*4+1], e2 = eattr[wid*4+2], e3 = eattr[wid*4+3];
    int c0 = lane * 8;
    #pragma unroll
    for (int i = 0; i < 8; ++i) {
        int c = c0 + i;
        float ev = e0*sWe[c] + e1*sWe[HC+c] + e2*sWe[2*HC+c] + e3*sWe[3*HC+c];
        atomicAdd(&out[(size_t)d * HC + c], (V[(size_t)s * HC + c] + ev) * a);
    }
}

// ---------------------------------------------------------------- pooling/head

__global__ void pool_cnt(const int* __restrict__ batch, float* __restrict__ cnt, int n) {
    int i = blockIdx.x * blockDim.x + threadIdx.x;
    if (i < n) atomicAdd(&cnt[batch[i]], 1.0f);
}

__global__ void pool_accum(const float* __restrict__ x, const int* __restrict__ batch,
                           float* __restrict__ pooled, int n) {
    int i = blockIdx.x * blockDim.x + threadIdx.x;
    if (i >= n) return;
    int node = i >> 8;
    atomicAdd(&pooled[(size_t)batch[node] * HC + (i & 255)], x[i]);
}

__global__ void head_kernel(const float* __restrict__ pooled, const float* __restrict__ cnt,
                            const float* __restrict__ Wh, const float* __restrict__ bh,
                            float* __restrict__ out) {
    int i = blockIdx.x * blockDim.x + threadIdx.x;
    if (i >= NG * NT) return;
    int g = i / NT, t = i % NT;
    float inv = 1.0f / fmaxf(cnt[g], 1.0f);
    float sum = bh[t];
    for (int c = 0; c < HC; ++c)
        sum += pooled[(size_t)g * HC + c] * inv * Wh[c * NT + t];
    out[i] = sum;
}

// ---------------------------------------------------------------- launch

extern "C" void kernel_launch(void* const* d_in, const int* in_sizes, int n_in,
                              void* d_out, int out_size, void* d_ws, size_t ws_size,
                              hipStream_t stream) {
    (void)in_sizes; (void)n_in; (void)out_size; (void)ws_size;

    const float* x     = (const float*)d_in[0];
    const int*   src   = (const int*)d_in[1];           // edge_index[0]
    const int*   dst   = ((const int*)d_in[1]) + NE;    // edge_index[1]
    const float* eattr = (const float*)d_in[2];
    const int*   batch = (const int*)d_in[3];
    // params flattened: 3 layers x (Wq,bq,Wk,bk,Wv,bv,We,Ws,bs), then (Wh,bh)
    const float* Wh = (const float*)d_in[4 + 9 * 3];
    const float* bh = (const float*)d_in[4 + 9 * 3 + 1];
    float* outG = (float*)d_out;

    // ---- workspace partition (~99 MB)
    char* base = (char*)d_ws;
    size_t off = 0;
    auto alloc = [&](size_t bytes) { void* p = base + off; off = (off + bytes + 255) & ~(size_t)255; return p; };
    float*  Qf     = (float*)alloc((size_t)NN * HC * 4);
    float*  Kf     = (float*)alloc((size_t)NN * HC * 4);
    float*  Vf     = (float*)alloc((size_t)NN * HC * 4);
    float*  Outf   = (float*)alloc((size_t)NN * HC * 4);
    __bf16* xb     = (__bf16*)alloc((size_t)NN * HC * 2);
    float*  alphaB = (float*)alloc((size_t)NE * NH * 4);
    float*  amax   = (float*)alloc((size_t)NN * NH * 4);
    float*  denom  = (float*)alloc((size_t)NN * NH * 4);
    float*  pooled = (float*)alloc((size_t)NG * HC * 4);
    float*  cnt    = (float*)alloc((size_t)NG * 4);
    __bf16* wswQ   = (__bf16*)alloc((size_t)HC * HC * 2);
    __bf16* wswK   = (__bf16*)alloc((size_t)HC * HC * 2);
    __bf16* wswV   = (__bf16*)alloc((size_t)HC * HC * 2);
    __bf16* wswS   = (__bf16*)alloc((size_t)HC * HC * 2);

    const int TB = 256;
    const int rowTiles   = NN / 16;                 // 1250
    const int edgeBlocks = NE / (TB / 32);          // 40000 (one wave per edge)

    // input features -> bf16 (ld = 64)
    cvt_f32_bf16<<<(NN * DIN + TB - 1) / TB, TB, 0, stream>>>(x, xb, NN * DIN);

    for (int l = 0; l < 3; ++l) {
        const int pbase = 4 + 9 * l;
        const float* Wq = (const float*)d_in[pbase + 0];
        const float* bq = (const float*)d_in[pbase + 1];
        const float* Wk = (const float*)d_in[pbase + 2];
        const float* bk = (const float*)d_in[pbase + 3];
        const float* Wv = (const float*)d_in[pbase + 4];
        const float* bv = (const float*)d_in[pbase + 5];
        const float* We = (const float*)d_in[pbase + 6];
        const float* Ws = (const float*)d_in[pbase + 7];
        const float* bs = (const float*)d_in[pbase + 8];
        const int Kdim = (l == 0) ? DIN : HC;
        const int ldx  = Kdim;
        const int swB  = (Kdim * HC + TB - 1) / TB;

        swizzle_w<<<swB, TB, 0, stream>>>(Wq, wswQ, Kdim);
        swizzle_w<<<swB, TB, 0, stream>>>(Wk, wswK, Kdim);
        swizzle_w<<<swB, TB, 0, stream>>>(Wv, wswV, Kdim);
        swizzle_w<<<swB, TB, 0, stream>>>(Ws, wswS, Kdim);

        gemm_wmma_bf16<<<rowTiles, 128, 0, stream>>>(xb, ldx, Kdim, wswQ, bq, Qf);
        gemm_wmma_bf16<<<rowTiles, 128, 0, stream>>>(xb, ldx, Kdim, wswK, bk, Kf);
        gemm_wmma_bf16<<<rowTiles, 128, 0, stream>>>(xb, ldx, Kdim, wswV, bv, Vf);
        gemm_wmma_bf16<<<rowTiles, 128, 0, stream>>>(xb, ldx, Kdim, wswS, bs, Outf); // skip path

        fill_f32<<<(NN * NH + TB - 1) / TB, TB, 0, stream>>>(amax, -INFINITY, NN * NH);
        fill_f32<<<(NN * NH + TB - 1) / TB, TB, 0, stream>>>(denom, 0.0f, NN * NH);

        edge_alpha  <<<edgeBlocks, TB, 0, stream>>>(Qf, Kf, eattr, We, src, dst, alphaB, amax);
        edge_exp    <<<(NE * NH + TB - 1) / TB, TB, 0, stream>>>(dst, amax, alphaB, denom, NE * NH);
        edge_scatter<<<edgeBlocks, TB, 0, stream>>>(Vf, eattr, We, src, dst, alphaB, denom, Outf);

        if (l < 2)  // relu + next-layer bf16 activations
            relu_cvt_f32_bf16<<<(NN * HC + TB - 1) / TB, TB, 0, stream>>>(Outf, xb, NN * HC);
    }

    // global mean pool + task heads
    fill_f32<<<(NG * HC + TB - 1) / TB, TB, 0, stream>>>(pooled, 0.0f, NG * HC);
    fill_f32<<<(NG + TB - 1) / TB, TB, 0, stream>>>(cnt, 0.0f, NG);
    pool_cnt  <<<(NN + TB - 1) / TB, TB, 0, stream>>>(batch, cnt, NN);
    pool_accum<<<(NN * HC + TB - 1) / TB, TB, 0, stream>>>(Outf, batch, pooled, NN * HC);
    head_kernel<<<(NG * NT + TB - 1) / TB, TB, 0, stream>>>(pooled, cnt, Wh, bh, outG);
}
